// GAEncoder_20444044329055
// MI455X (gfx1250) — matrix operand order repaired and checked by reading, not compiled, not run on hardware
//
#include <hip/hip_runtime.h>
#include <hip/hip_bf16.h>
#include <stdint.h>

#define N_NODES 50000
#define N_EDGES 640000
#define DIM_IN  128
#define DIM_H   128
#define DIM_OUT 64
#define N_BR    3
#define GK      128        // K is 128 for both layers
#define MTB     4          // M-tiles per block (amortizes B-register reuse)

typedef __attribute__((ext_vector_type(16))) __bf16 v16bf;
typedef __attribute__((ext_vector_type(8)))  float  v8f;

// RNE f32->bf16 (used only in one-time weight conversion)
static __device__ __forceinline__ __bf16 f2bf(float f) {
  union { float f; uint32_t u; } x; x.f = f;
  uint32_t r = x.u + 0x7FFFu + ((x.u >> 16) & 1u);
  union { uint16_t s; __bf16 b; } y; y.s = (uint16_t)(r >> 16);
  return y.b;
}

// One v_perm_b32: pack {hi[31:16], lo[31:16]} = two bf16 (truncation) in a single VALU op
static __device__ __forceinline__ uint32_t pack_bf16x2(float lo, float hi) {
  union { float f; uint32_t u; } a, b; a.f = lo; b.f = hi;
  return __builtin_amdgcn_perm(b.u, a.u, 0x07060302u);
}

static __device__ __forceinline__ float4 biasrelu4(float4 q, float4 b) {
  q.x = fmaxf(q.x + b.x, 0.f); q.y = fmaxf(q.y + b.y, 0.f);
  q.z = fmaxf(q.z + b.z, 0.f); q.w = fmaxf(q.w + b.w, 0.f);
  return q;
}

// ---------------- degree / normalization ----------------
__global__ void k_fill1(float* __restrict__ p, int n) {
  int i = blockIdx.x * blockDim.x + threadIdx.x;
  if (i < n) p[i] = 1.0f;                    // self-loop contributes 1 to degree
}

__global__ void k_count(const long long* __restrict__ dst, float* __restrict__ deg, int nE) {
  int e = blockIdx.x * blockDim.x + threadIdx.x;
  if (e < nE) atomicAdd(&deg[(int)dst[e]], 1.0f);
}

__global__ void k_rsqrt(float* __restrict__ p, int n) {
  int i = blockIdx.x * blockDim.x + threadIdx.x;
  if (i < n) p[i] = rsqrtf(p[i]);            // deg >= 1 always (self-loops)
}

// ---------------- one-time weight transpose+convert: Wt[c*K + k] = bf16(W[k*Nc + c]) ----------------
__global__ void k_convW(const float* __restrict__ W, __bf16* __restrict__ Wt, int K, int Nc) {
  int idx = blockIdx.x * blockDim.x + threadIdx.x;
  if (idx >= K * Nc) return;
  int k = idx / Nc, c = idx % Nc;
  Wt[(size_t)c * K + k] = f2bf(W[idx]);
}

// ---------------- WMMA bf16 GEMM: C[M,Nc] = op(A[M,GK]) @ B, B pre-converted as Bt[Nc][GK] bf16 ----
// op(A)[m][k] = FUSE ? relu(A[m][k] + biasK[k]) : A[m][k]
// block = (Nc/16) waves; wave w -> column tile w; each block walks MTB M-tiles.
template <bool FUSE>
__global__ void __launch_bounds__(256)
k_gemm_bf16(const float* __restrict__ A, const __bf16* __restrict__ Bt,
            const float* __restrict__ biasK, float* __restrict__ C,
            int Mtiles, int Nc)
{
  const int lane  = threadIdx.x & 31;
  const int wave  = threadIdx.x >> 5;
  const int lmod  = lane & 15;   // row (A) / col (B,C) within tile
  const int lhalf = lane >> 4;   // K-half selector (A,B) / row-half (C)
  const int col0  = wave * 16;

  // ---- hoist B fragments for whole K=128 (4 chunks): 32 contiguous bytes per chunk per lane ----
  v16bf breg[4];
  {
    const v16bf* __restrict__ bp =
        (const v16bf*)(Bt + (size_t)(col0 + lmod) * GK);   // 32B-aligned (col*256B)
#pragma unroll
    for (int c = 0; c < 4; ++c)
      breg[c] = bp[2 * c + lhalf];                          // K = 32c + 16*lhalf + [0..16)
  }

  // ---- hoist bias fragments for the K indices this lane touches ----
  // chunk c, lane half h: K = 32c + 16h + [0..16)
  float4 bb[4][4];
  if (FUSE) {
#pragma unroll
    for (int c = 0; c < 4; ++c) {
      const float* p0 = biasK + 32 * c + 16 * lhalf;
      bb[c][0] = *(const float4*)(p0);          // lo  +0..3
      bb[c][1] = *(const float4*)(p0 + 4);      // lo  +4..7
      bb[c][2] = *(const float4*)(p0 + 8);      // hi  +8..11
      bb[c][3] = *(const float4*)(p0 + 12);     // hi  +12..15
    }
  }

  for (int t = 0; t < MTB; ++t) {
    const int mtile = blockIdx.x * MTB + t;
    if (mtile >= Mtiles) break;                       // wave-uniform: EXEC stays all-ones
    const int row0 = mtile * 16;
    const float* __restrict__ arow = A + (size_t)(row0 + lmod) * GK;
    if (t + 1 < MTB)
      __builtin_prefetch(arow + 16 * GK, 0, 1);       // next M-tile -> global_prefetch_b8

    v8f acc = {};
#pragma unroll
    for (int c = 0; c < 4; ++c) {
      const int kk = 32 * c;
      // A fragment: lane row = row0+lmod; K = kk+16h + [0..16) (q0..q3)
      float4 q0 = *(const float4*)(arow + kk + 16 * lhalf);
      float4 q1 = *(const float4*)(arow + kk + 16 * lhalf + 4);
      float4 q2 = *(const float4*)(arow + kk + 16 * lhalf + 8);
      float4 q3 = *(const float4*)(arow + kk + 16 * lhalf + 12);
      if (FUSE) {
        q0 = biasrelu4(q0, bb[c][0]);
        q1 = biasrelu4(q1, bb[c][1]);
        q2 = biasrelu4(q2, bb[c][2]);
        q3 = biasrelu4(q3, bb[c][3]);
      }
      union { uint32_t u[8]; v16bf v; } af;
      af.u[0] = pack_bf16x2(q0.x, q0.y); af.u[1] = pack_bf16x2(q0.z, q0.w);
      af.u[2] = pack_bf16x2(q1.x, q1.y); af.u[3] = pack_bf16x2(q1.z, q1.w);
      af.u[4] = pack_bf16x2(q2.x, q2.y); af.u[5] = pack_bf16x2(q2.z, q2.w);
      af.u[6] = pack_bf16x2(q3.x, q3.y); af.u[7] = pack_bf16x2(q3.z, q3.w);

      acc = __builtin_amdgcn_wmma_f32_16x16x32_bf16(
          false, af.v, false, breg[c], (short)0, acc, false, false);
    }

    // C 16x16 f32: VGPR r -> row = row0 + 8*lhalf + r, col = col0 + lmod
    float* crow = C + (size_t)(row0 + 8 * lhalf) * Nc + (col0 + lmod);
#pragma unroll
    for (int r = 0; r < 8; ++r)
      crow[(size_t)r * Nc] = acc[r];
  }
}

// ---------------- self-loop init: Out = dinv[node]^2 * T  (covers edge (i,i)) ----------------
__global__ void k_selfloop_init(const float* __restrict__ T, const float* __restrict__ dinv,
                                float* __restrict__ Out, int n, int F)
{
  size_t idx = (size_t)blockIdx.x * blockDim.x + threadIdx.x;
  size_t total = (size_t)N_BR * n * F;
  if (idx >= total) return;
  int node = (int)((idx / (size_t)F) % (size_t)n);
  float di = dinv[node];
  Out[idx] = di * di * T[idx];
}

// ---------------- edge scatter: Out[b][dst] += dinv[src]*dinv[dst] * T[b][src] ----------------
// one wave per edge; lane stride 1 over features -> fully coalesced loads & atomics
template <int F>
__global__ void __launch_bounds__(256)
k_scatter(const long long* __restrict__ src, const long long* __restrict__ dst,
          const float* __restrict__ dinv, const float* __restrict__ T,
          float* __restrict__ Out, int nE, int n)
{
  const int lane = threadIdx.x & 31;
  const int e    = blockIdx.x * (blockDim.x >> 5) + (threadIdx.x >> 5);
  if (e >= nE) return;
  const int b = blockIdx.y;
  const int s = (int)src[e];
  const int d = (int)dst[e];
  const float w = dinv[s] * dinv[d];
  const float* tp = T   + ((size_t)b * n + s) * F;
  float*       op = Out + ((size_t)b * n + d) * F;
#pragma unroll
  for (int j = 0; j < F / 32; ++j) {
    const int c = lane + 32 * j;
    atomicAdd(&op[c], w * tp[c]);   // global_atomic_add_f32, no return
  }
}

// ---------------- final: out = (sum of 3 branches)/3 + b2 ----------------
__global__ void k_final(const float* __restrict__ Agg, const float* __restrict__ b2,
                        float* __restrict__ out, int n, int F)
{
  int idx = blockIdx.x * blockDim.x + threadIdx.x;
  if (idx >= n * F) return;
  size_t stride = (size_t)n * F;
  float v = Agg[idx] + Agg[idx + stride] + Agg[idx + 2 * stride];
  out[idx] = v * (1.0f / 3.0f) + b2[idx % F];
}

extern "C" void kernel_launch(void* const* d_in, const int* in_sizes, int n_in,
                              void* d_out, int out_size, void* d_ws, size_t ws_size,
                              hipStream_t stream)
{
  const float* X[3] = { (const float*)d_in[0], (const float*)d_in[1], (const float*)d_in[2] };
  const long long* eidx = (const long long*)d_in[3];      // int64 per reference
  const long long* esrc = eidx;                           // edge_index[0]
  const long long* edst = eidx + N_EDGES;                 // edge_index[1]
  const float* W1 = (const float*)d_in[4];
  const float* b1 = (const float*)d_in[5];
  const float* W2 = (const float*)d_in[6];
  const float* b2 = (const float*)d_in[7];
  float* out = (float*)d_out;

  // workspace: dinv | W1t bf16 | W2t bf16 | bufA (150000x128 f32) | bufB (150000x128 f32)
  char* ws = (char*)d_ws;
  float* dinv = (float*)ws;
  size_t o = ((size_t)N_NODES * sizeof(float) + 255) & ~(size_t)255;
  __bf16* W1t = (__bf16*)(ws + o);            o += ((size_t)DIM_H   * GK * 2 + 255) & ~(size_t)255;
  __bf16* W2t = (__bf16*)(ws + o);            o += ((size_t)DIM_OUT * GK * 2 + 255) & ~(size_t)255;
  float* bufA = (float*)(ws + o);
  size_t szA  = (size_t)N_BR * N_NODES * DIM_H * sizeof(float);   // 76.8 MB
  float* bufB = (float*)(ws + o + szA);
  (void)in_sizes; (void)n_in; (void)out_size; (void)ws_size;

  // 1) degree -> dinv = deg^{-1/2}; weights -> transposed bf16
  k_fill1 <<<(N_NODES + 255) / 256, 256, 0, stream>>>(dinv, N_NODES);
  k_count <<<(N_EDGES + 255) / 256, 256, 0, stream>>>(edst, dinv, N_EDGES);
  k_rsqrt <<<(N_NODES + 255) / 256, 256, 0, stream>>>(dinv, N_NODES);
  k_convW <<<(GK * DIM_H   + 255) / 256, 256, 0, stream>>>(W1, W1t, GK, DIM_H);
  k_convW <<<(GK * DIM_OUT + 255) / 256, 256, 0, stream>>>(W2, W2t, GK, DIM_OUT);

  // 2) layer-1 GEMM per branch: bufA[b] = X_b @ W1   (Mtiles=3125, Nc=128, 8 waves)
  {
    int Mtiles = N_NODES / 16;
    int blocks = (Mtiles + MTB - 1) / MTB;
    for (int b = 0; b < N_BR; ++b)
      k_gemm_bf16<false><<<blocks, (DIM_H / 16) * 32, 0, stream>>>(
          X[b], W1t, nullptr, bufA + (size_t)b * N_NODES * DIM_H, Mtiles, DIM_H);
  }

  // 3) aggregate layer 1 into bufB (self-loops init + edge atomics)
  {
    size_t tot = (size_t)N_BR * N_NODES * DIM_H;
    k_selfloop_init<<<(unsigned)((tot + 255) / 256), 256, 0, stream>>>(bufA, dinv, bufB, N_NODES, DIM_H);
    dim3 g((N_EDGES + 7) / 8, N_BR);
    k_scatter<DIM_H><<<g, 256, 0, stream>>>(esrc, edst, dinv, bufA, bufB, N_EDGES, N_NODES);
  }

  // 4) layer-2 GEMM, fused bias1+ReLU on A: bufA = relu(bufB + b1) @ W2  (Mtiles=9375, Nc=64, 4 waves)
  {
    int Mtiles = (N_BR * N_NODES) / 16;
    int blocks = (Mtiles + MTB - 1) / MTB;
    k_gemm_bf16<true><<<blocks, (DIM_OUT / 16) * 32, 0, stream>>>(
        bufB, W2t, b1, bufA, Mtiles, DIM_OUT);
  }

  // 5) aggregate layer 2 into bufB
  {
    size_t tot = (size_t)N_BR * N_NODES * DIM_OUT;
    k_selfloop_init<<<(unsigned)((tot + 255) / 256), 256, 0, stream>>>(bufA, dinv, bufB, N_NODES, DIM_OUT);
    dim3 g((N_EDGES + 7) / 8, N_BR);
    k_scatter<DIM_OUT><<<g, 256, 0, stream>>>(esrc, edst, dinv, bufA, bufB, N_EDGES, N_NODES);
  }

  // 6) average branches + bias2
  k_final<<<(N_NODES * DIM_OUT + 255) / 256, 256, 0, stream>>>(bufB, b2, out, N_NODES, DIM_OUT);
}
